// VanillaTransformerModel_54984171323935
// MI455X (gfx1250) — compile-verified, matrix operand order checked by reading, not run on hardware
//
#include <hip/hip_runtime.h>
#include <hip/hip_bf16.h>

// ---------------------------------------------------------------------------
// Types for CDNA5 WMMA (wave32): D(16x16 f32) = A(16x32 bf16) x B(32x16 bf16) + C
// ---------------------------------------------------------------------------
typedef __bf16 bf16;
typedef __attribute__((ext_vector_type(16))) __bf16 v16bf;
typedef __attribute__((ext_vector_type(8)))  float  v8f;
typedef int v4i __attribute__((vector_size(16)));

union FragB { v16bf v; uint4 q[2]; };

#if __has_builtin(__builtin_amdgcn_global_load_async_to_lds_b128) && \
    __has_builtin(__builtin_amdgcn_s_wait_asynccnt)
#define ASYNC_LDS 1
#define GPTR(p) ((v4i*)(p))
#define LPTR(p) ((v4i*)(p))
#else
#define ASYNC_LDS 0
#endif

// float -> bf16, round to nearest even
__device__ __forceinline__ bf16 f2b(float f) {
    unsigned u = __float_as_uint(f);
    unsigned r = (u + 0x7FFFu + ((u >> 16) & 1u)) >> 16;
    unsigned short s = (unsigned short)r;
    return __builtin_bit_cast(bf16, s);
}

// Load a 16x32 bf16 fragment (A-layout; B-layout is identical when loading
// the N x K operand row-major) from a row-major tile with stride `stride`
// (bf16 elements; multiple of 8 for 16B alignment).
// ISA 7.12.2: lane L -> row r=L%16, g=L/16; elems 0..7 = cols 8g+0..7,
// elems 8..15 = cols 16+8g+0..7.
__device__ __forceinline__ v16bf load_frag(const bf16* base, int stride) {
    const int lane = threadIdx.x & 31;
    const int r = lane & 15, g = lane >> 4;
    FragB f;
    f.q[0] = *(const uint4*)(base + r * stride + 8 * g);
    f.q[1] = *(const uint4*)(base + r * stride + 16 + 8 * g);
    return f.v;
}

__device__ __forceinline__ v8f wmma_bf16(v16bf a, v16bf b, v8f c) {
    return __builtin_amdgcn_wmma_f32_16x16x32_bf16(false, a, false, b,
                                                   (short)0, c, false, false);
}

// ---------------------------------------------------------------------------
// fp32 -> bf16 weight conversion
// ---------------------------------------------------------------------------
__global__ __launch_bounds__(256) void cvt_kernel(const float* __restrict__ in,
                                                  bf16* __restrict__ out,
                                                  long long n) {
    long long i = ((long long)blockIdx.x * 256 + threadIdx.x) * 4;
    if (i >= n) return;
    float4 v = *(const float4*)(in + i);
    out[i + 0] = f2b(v.x);
    out[i + 1] = f2b(v.y);
    out[i + 2] = f2b(v.z);
    out[i + 3] = f2b(v.w);
}

// ---------------------------------------------------------------------------
// Embedding: x[t,c] = wte[ids[t],c] + wpe[t,c]   (T=2048, C=1024)
// ---------------------------------------------------------------------------
__global__ __launch_bounds__(256) void embed_kernel(const int* __restrict__ ids,
                                                    const float* __restrict__ wte,
                                                    const float* __restrict__ wpe,
                                                    float* __restrict__ x) {
    int idx = blockIdx.x * 256 + threadIdx.x;   // over T*C
    int t = idx >> 10;
    int c = idx & 1023;
    x[idx] = wte[(size_t)ids[t] * 1024 + c] + wpe[idx];
}

// ---------------------------------------------------------------------------
// LayerNorm over C=1024, one block (256 thr) per row; bf16 output
// ---------------------------------------------------------------------------
__global__ __launch_bounds__(256) void ln_kernel(const float* __restrict__ x,
                                                 const float* __restrict__ w,
                                                 const float* __restrict__ b,
                                                 bf16* __restrict__ out) {
    __shared__ float rs[256], rs2[256];
    const int row = blockIdx.x, tid = threadIdx.x;
    const float* xr = x + (size_t)row * 1024;
    float4 v = *(const float4*)(xr + tid * 4);
    float s  = v.x + v.y + v.z + v.w;
    float s2 = v.x * v.x + v.y * v.y + v.z * v.z + v.w * v.w;
    rs[tid] = s; rs2[tid] = s2;
    __syncthreads();
    for (int off = 128; off > 0; off >>= 1) {
        if (tid < off) { rs[tid] += rs[tid + off]; rs2[tid] += rs2[tid + off]; }
        __syncthreads();
    }
    float mu  = rs[0] * (1.0f / 1024.0f);
    float var = rs2[0] * (1.0f / 1024.0f) - mu * mu;
    float rr  = rsqrtf(var + 1e-5f);
    bf16* orow = out + (size_t)row * 1024;
    float4 wv = *(const float4*)(w + tid * 4);
    float4 bv = *(const float4*)(b + tid * 4);
    orow[tid * 4 + 0] = f2b((v.x - mu) * rr * wv.x + bv.x);
    orow[tid * 4 + 1] = f2b((v.y - mu) * rr * wv.y + bv.y);
    orow[tid * 4 + 2] = f2b((v.z - mu) * rr * wv.z + bv.z);
    orow[tid * 4 + 3] = f2b((v.w - mu) * rr * wv.w + bv.w);
}

// ---------------------------------------------------------------------------
// GEMM: out[M,N] = A[M,K](bf16,rm) x W[N,K](bf16,rm)^T [+bias][+res][gelu]
// Block tile 128x128, 8 waves (4x2), wave tile 32x64, K step 32.
// Double-buffered LDS, one barrier per K step; async HBM->LDS when available.
// ---------------------------------------------------------------------------
#define LDST 40  // LDS row stride in bf16 (32 data + 8 pad), 80B rows

template <bool HAS_BIAS, bool HAS_RES, bool DO_GELU, bool OUT_F, bool OUT_B>
__global__ __launch_bounds__(256) void gemm_bf16_kernel(
    const bf16* __restrict__ A, const bf16* __restrict__ W,
    const float* __restrict__ bias, const float* __restrict__ res,
    float* __restrict__ outF, bf16* __restrict__ outB,
    int N, int K) {
    __shared__ bf16 As[2][128 * LDST];
    __shared__ bf16 Bs[2][128 * LDST];

    const int tid  = threadIdx.x;
    const int lane = tid & 31, wave = tid >> 5;
    const int wm = wave >> 1, wn = wave & 1;          // 4 x 2 wave grid
    const int m0 = blockIdx.y * 128, n0 = blockIdx.x * 128;

    v8f acc[2][4];
#pragma unroll
    for (int i = 0; i < 2; ++i)
#pragma unroll
        for (int j = 0; j < 4; ++j) acc[i][j] = (v8f){0, 0, 0, 0, 0, 0, 0, 0};

    // cooperative staging: 256 threads, 2 per row, 32B each
    const int lrow = tid >> 1;        // 0..127
    const int lhalf = (tid & 1) * 16; // element offset within 32-wide K slab
    const bf16* aG = A + (size_t)(m0 + lrow) * K + lhalf;
    const bf16* wG = W + (size_t)(n0 + lrow) * K + lhalf;
    const int lds_off = lrow * LDST + lhalf;

    int buf = 0;
    for (int k0 = 0; k0 < K; k0 += 32, buf ^= 1) {
        // prefetch two K-steps ahead toward L2 (global_prefetch_b8)
        if (k0 + 64 < K) {
            __builtin_prefetch(aG + k0 + 64, 0, 1);
            __builtin_prefetch(wG + k0 + 64, 0, 1);
        }
#if ASYNC_LDS
        __builtin_amdgcn_global_load_async_to_lds_b128(
            GPTR(aG + k0),     LPTR(&As[buf][lds_off]),     0, 0);
        __builtin_amdgcn_global_load_async_to_lds_b128(
            GPTR(aG + k0 + 8), LPTR(&As[buf][lds_off + 8]), 0, 0);
        __builtin_amdgcn_global_load_async_to_lds_b128(
            GPTR(wG + k0),     LPTR(&Bs[buf][lds_off]),     0, 0);
        __builtin_amdgcn_global_load_async_to_lds_b128(
            GPTR(wG + k0 + 8), LPTR(&Bs[buf][lds_off + 8]), 0, 0);
        __builtin_amdgcn_s_wait_asynccnt(0);
#else
        uint4 a0 = *(const uint4*)(aG + k0);
        uint4 a1 = *(const uint4*)(aG + k0 + 8);
        uint4 b0 = *(const uint4*)(wG + k0);
        uint4 b1 = *(const uint4*)(wG + k0 + 8);
        *(uint4*)(&As[buf][lds_off])     = a0;
        *(uint4*)(&As[buf][lds_off + 8]) = a1;
        *(uint4*)(&Bs[buf][lds_off])     = b0;
        *(uint4*)(&Bs[buf][lds_off + 8]) = b1;
#endif
        __syncthreads();

        v16bf af[2], bfr[4];
#pragma unroll
        for (int i = 0; i < 2; ++i)
            af[i] = load_frag(&As[buf][(wm * 32 + i * 16) * LDST], LDST);
#pragma unroll
        for (int j = 0; j < 4; ++j)
            bfr[j] = load_frag(&Bs[buf][(wn * 64 + j * 16) * LDST], LDST);
#pragma unroll
        for (int i = 0; i < 2; ++i)
#pragma unroll
            for (int j = 0; j < 4; ++j)
                acc[i][j] = wmma_bf16(af[i], bfr[j], acc[i][j]);
    }

    // epilogue: lane holds col n=lane%16, rows e+8g  (ISA 7.12.2 C layout)
    const int g = lane >> 4, cn = lane & 15;
#pragma unroll
    for (int i = 0; i < 2; ++i) {
#pragma unroll
        for (int j = 0; j < 4; ++j) {
            const int colg = n0 + wn * 64 + j * 16 + cn;
            float bv = 0.0f;
            if (HAS_BIAS) bv = bias[colg];
#pragma unroll
            for (int e = 0; e < 8; ++e) {
                const int rowg = m0 + wm * 32 + i * 16 + 8 * g + e;
                const size_t idx = (size_t)rowg * N + colg;
                float v = acc[i][j][e] + bv;
                if (HAS_RES)  v += res[idx];
                if (DO_GELU)  v = 0.5f * v * (1.0f + erff(v * 0.70710678118f));
                if (OUT_F)    outF[idx] = v;
                if (OUT_B)    outB[idx] = f2b(v);
            }
        }
    }
}

// ---------------------------------------------------------------------------
// Causal flash attention (transposed formulation), H=16, D=64, T=2048.
// qkv: (T, 3C) bf16 with q|k|v sections; y: (T, C) bf16 output.
// Block: 8 waves, head = blockIdx.y, queries [blockIdx.x*128, +128),
// wave w owns 16 query rows. Key blocks of 32, double-buffered in LDS
// (V stored transposed). S^T = K.Q^T (both operands load naturally from
// row-major memory); per-column online softmax (8 in-lane + shfl_xor(16));
// P^T C-frags repack in-lane into B-frags; O^T += V^T.P^T.
// ---------------------------------------------------------------------------
__global__ __launch_bounds__(256) void attn_kernel(const bf16* __restrict__ qkv,
                                                   bf16* __restrict__ y) {
    __shared__ bf16 Kt[2][32 * 72];  // 32 keys x 64 d (stride 72)
    __shared__ bf16 Vt[2][64 * 40];  // 64 d x 32 keys (stride 40), transposed

    const int tid = threadIdx.x, lane = tid & 31, wave = tid >> 5;
    const int h  = blockIdx.y;
    const int q0 = blockIdx.x * 128;
    const int qw0 = q0 + wave * 16;
    const int r = lane & 15, g = lane >> 4;

    // Q B-fragments (B-layout loads naturally from row-major Q[16 x 64])
    v16bf qf[2];
    {
        const bf16* qbase = qkv + (size_t)(qw0 + r) * 3072 + h * 64;
#pragma unroll
        for (int ks = 0; ks < 2; ++ks) {
            FragB f;
            f.q[0] = *(const uint4*)(qbase + ks * 32 + 8 * g);
            f.q[1] = *(const uint4*)(qbase + ks * 32 + 16 + 8 * g);
            qf[ks] = f.v;
        }
    }

    v8f ot[4];
#pragma unroll
    for (int md = 0; md < 4; ++md) ot[md] = (v8f){0, 0, 0, 0, 0, 0, 0, 0};
    float mi = -INFINITY, li = 0.0f;
    const float scale = 0.125f;  // 1/sqrt(64)

    const int srow = tid >> 3;   // staging: key row 0..31
    const int schk = tid & 7;    // 8-elem chunk within d=64
    const int nkb = (q0 + 128) >> 5;
    const int qg = qw0 + r;      // this lane's query (column of S^T)
    // wave-uniform causal bound, forced scalar so EXEC stays all-1 for WMMA
    const int qwave_hi = __builtin_amdgcn_readfirstlane(qw0 + 15);

    const bf16* kgb = qkv + (size_t)srow * 3072 + 1024 + h * 64 + schk * 8;
    const bf16* vgb = qkv + (size_t)srow * 3072 + 2048 + h * 64 + schk * 8;

    int buf = 0;
    for (int j = 0; j < nkb; ++j, buf ^= 1) {
        const int kb0 = j * 32;
        if (j + 1 < nkb) {
            __builtin_prefetch(kgb + (size_t)(kb0 + 32) * 3072, 0, 1);
            __builtin_prefetch(vgb + (size_t)(kb0 + 32) * 3072, 0, 1);
        }
        uint4 v4 = *(const uint4*)(vgb + (size_t)kb0 * 3072);
#if ASYNC_LDS
        __builtin_amdgcn_global_load_async_to_lds_b128(
            GPTR(kgb + (size_t)kb0 * 3072),
            LPTR(&Kt[buf][srow * 72 + schk * 8]), 0, 0);
#else
        uint4 k4 = *(const uint4*)(kgb + (size_t)kb0 * 3072);
        *(uint4*)(&Kt[buf][srow * 72 + schk * 8]) = k4;
#endif
        const bf16* vp = (const bf16*)&v4;
#pragma unroll
        for (int e = 0; e < 8; ++e) Vt[buf][(schk * 8 + e) * 40 + srow] = vp[e];
#if ASYNC_LDS
        __builtin_amdgcn_s_wait_asynccnt(0);
#endif
        __syncthreads();

        if (kb0 <= qwave_hi) {  // scalar branch; EXEC all-1 inside
            // batch all K-fragment loads, then the 4 WMMAs
            v16bf kf[4];
            kf[0] = load_frag(&Kt[buf][0], 72);            // keys 0-15,  d 0-31
            kf[1] = load_frag(&Kt[buf][32], 72);           // keys 0-15,  d 32-63
            kf[2] = load_frag(&Kt[buf][16 * 72], 72);      // keys 16-31, d 0-31
            kf[3] = load_frag(&Kt[buf][16 * 72 + 32], 72); // keys 16-31, d 32-63
            v8f s0 = (v8f){0, 0, 0, 0, 0, 0, 0, 0};
            v8f s1 = (v8f){0, 0, 0, 0, 0, 0, 0, 0};
            s0 = wmma_bf16(kf[0], qf[0], s0);
            s0 = wmma_bf16(kf[1], qf[1], s0);
            s1 = wmma_bf16(kf[2], qf[0], s1);
            s1 = wmma_bf16(kf[3], qf[1], s1);

            // scale + causal mask; online softmax over this lane's column qg
            float p0[8], p1[8];
            float mnew = mi;
#pragma unroll
            for (int e = 0; e < 8; ++e) {
                int k0e = kb0 + 8 * g + e;
                float v = (k0e <= qg) ? s0[e] * scale : -INFINITY;
                p0[e] = v; mnew = fmaxf(mnew, v);
                int k1e = kb0 + 16 + 8 * g + e;
                v = (k1e <= qg) ? s1[e] * scale : -INFINITY;
                p1[e] = v; mnew = fmaxf(mnew, v);
            }
            mnew = fmaxf(mnew, __shfl_xor(mnew, 16, 32));
            float alpha = __expf(mi - mnew);
            float ls = 0.0f;
#pragma unroll
            for (int e = 0; e < 8; ++e) {
                p0[e] = __expf(p0[e] - mnew); ls += p0[e];
                p1[e] = __expf(p1[e] - mnew); ls += p1[e];
            }
            ls += __shfl_xor(ls, 16, 32);
            li = li * alpha + ls;
            mi = mnew;
#pragma unroll
            for (int md = 0; md < 4; ++md)
#pragma unroll
                for (int e = 0; e < 8; ++e) ot[md][e] *= alpha;

            // repack P^T C-frags -> one bf16 B-frag, entirely in-lane:
            // B elems 0..7 = keys 8g+0..7 (=s0 rows), 8..15 = keys 16+8g (=s1)
            FragB pf;
#pragma unroll
            for (int e = 0; e < 8; ++e) {
                pf.v[e]     = f2b(p0[e]);
                pf.v[8 + e] = f2b(p1[e]);
            }
            // O^T[d,q] += V^T[d,keys] x P^T[keys,q]; A = Vt rows (natural)
            v16bf vf[4];
#pragma unroll
            for (int md = 0; md < 4; ++md)
                vf[md] = load_frag(&Vt[buf][md * 16 * 40], 40);
#pragma unroll
            for (int md = 0; md < 4; ++md)
                ot[md] = wmma_bf16(vf[md], pf.v, ot[md]);
        }
    }

    // write y[q, h*64 + d] = O^T[d,q] / l
    float inv = 1.0f / li;
    bf16* yrow = y + (size_t)qg * 1024 + h * 64;
#pragma unroll
    for (int md = 0; md < 4; ++md)
#pragma unroll
        for (int e = 0; e < 8; ++e) {
            int d = md * 16 + 8 * g + e;
            yrow[d] = f2b(ot[md][e] * inv);
        }
}

// ---------------------------------------------------------------------------
// Host-side orchestration
// ---------------------------------------------------------------------------
extern "C" void kernel_launch(void* const* d_in, const int* in_sizes, int n_in,
                              void* d_out, int out_size, void* d_ws, size_t ws_size,
                              hipStream_t stream) {
    (void)in_sizes; (void)n_in; (void)out_size; (void)ws_size;
    const int T = 2048, C = 1024, Hn = 16, L = 4, V = 32000;

    const int*   ids    = (const int*)  d_in[0];
    const float* wte    = (const float*)d_in[1];
    const float* wpe    = (const float*)d_in[2];
    const float* ln1w   = (const float*)d_in[3];
    const float* ln1b   = (const float*)d_in[4];
    const float* attn_w = (const float*)d_in[5];
    const float* attn_b = (const float*)d_in[6];
    const float* proj_w = (const float*)d_in[7];
    const float* proj_b = (const float*)d_in[8];
    const float* ln2w   = (const float*)d_in[9];
    const float* ln2b   = (const float*)d_in[10];
    const float* fc_w   = (const float*)d_in[11];
    const float* fc_b   = (const float*)d_in[12];
    const float* fcp_w  = (const float*)d_in[13];
    const float* fcp_b  = (const float*)d_in[14];
    const float* lnfw   = (const float*)d_in[15];
    const float* lnfb   = (const float*)d_in[16];

    // workspace bump allocator
    char* p = (char*)d_ws;
    auto take = [&](size_t bytes) -> char* {
        char* r = p;
        p += (bytes + 255) & ~(size_t)255;
        return r;
    };
    float* Xa   = (float*)take((size_t)T * C * 4);
    float* Xb   = (float*)take((size_t)T * C * 4);
    bf16*  Hb   = (bf16*) take((size_t)T * C * 2);
    bf16*  QKVb = (bf16*) take((size_t)T * 3 * C * 2);
    bf16*  Yb   = (bf16*) take((size_t)T * C * 2);
    bf16*  Gb   = (bf16*) take((size_t)T * 4 * C * 2);
    bf16*  wteB  = (bf16*)take((size_t)V * C * 2);
    bf16*  attnB = (bf16*)take((size_t)L * 3 * C * C * 2);
    bf16*  projB = (bf16*)take((size_t)L * C * C * 2);
    bf16*  fcB   = (bf16*)take((size_t)L * 4 * C * C * 2);
    bf16*  fcpB  = (bf16*)take((size_t)L * 4 * C * C * 2);

    auto cvt = [&](const float* in, bf16* out, size_t n) {
        int blocks = (int)((n / 4 + 255) / 256);
        cvt_kernel<<<blocks, 256, 0, stream>>>(in, out, (long long)n);
    };
    cvt(wte,    wteB,  (size_t)V * C);
    cvt(attn_w, attnB, (size_t)L * 3 * C * C);
    cvt(proj_w, projB, (size_t)L * C * C);
    cvt(fc_w,   fcB,   (size_t)L * 4 * C * C);
    cvt(fcp_w,  fcpB,  (size_t)L * 4 * C * C);

    embed_kernel<<<(T * C) / 256, 256, 0, stream>>>(ids, wte, wpe, Xa);

    for (int l = 0; l < L; ++l) {
        ln_kernel<<<T, 256, 0, stream>>>(Xa, ln1w + l * C, ln1b + l * C, Hb);
        // QKV: bias, bf16 out
        gemm_bf16_kernel<true, false, false, false, true>
            <<<dim3(3 * C / 128, T / 128), 256, 0, stream>>>(
                Hb, attnB + (size_t)l * 3 * C * C, attn_b + l * 3 * C,
                nullptr, nullptr, QKVb, 3 * C, C);
        attn_kernel<<<dim3(T / 128, Hn), 256, 0, stream>>>(QKVb, Yb);
        // proj: bias + residual(Xa) -> f32 Xb
        gemm_bf16_kernel<true, true, false, true, false>
            <<<dim3(C / 128, T / 128), 256, 0, stream>>>(
                Yb, projB + (size_t)l * C * C, proj_b + l * C,
                Xa, Xb, nullptr, C, C);
        ln_kernel<<<T, 256, 0, stream>>>(Xb, ln2w + l * C, ln2b + l * C, Hb);
        // fc: bias + exact GELU -> bf16 Gb
        gemm_bf16_kernel<true, false, true, false, true>
            <<<dim3(4 * C / 128, T / 128), 256, 0, stream>>>(
                Hb, fcB + (size_t)l * 4 * C * C, fc_b + l * 4 * C,
                nullptr, nullptr, Gb, 4 * C, C);
        // fcp: bias + residual(Xb) -> f32 Xa
        gemm_bf16_kernel<true, true, false, true, false>
            <<<dim3(C / 128, T / 128), 256, 0, stream>>>(
                Gb, fcpB + (size_t)l * 4 * C * C, fcp_b + l * C,
                Xb, Xa, nullptr, C, 4 * C);
    }

    ln_kernel<<<T, 256, 0, stream>>>(Xa, lnfw, lnfb, Hb);
    // logits: no bias, f32 out
    gemm_bf16_kernel<false, false, false, true, false>
        <<<dim3(V / 128, T / 128), 256, 0, stream>>>(
            Hb, wteB, nullptr, nullptr, (float*)d_out, nullptr, V, C);
}
